// MultiHeadAttention_61813169324176
// MI455X (gfx1250) — matrix operand order repaired and checked by reading, not compiled, not run on hardware
//
#include <hip/hip_runtime.h>
#include <math.h>

typedef __attribute__((ext_vector_type(16))) __bf16 v16bf;
typedef __attribute__((ext_vector_type(8)))  float  v8f;
typedef __attribute__((ext_vector_type(4)))  int    v4i;

#if __has_builtin(__builtin_amdgcn_global_load_async_to_lds_b128)
#define ASYNC_LDS 1
#else
#define ASYNC_LDS 0
#endif

union BF16Frag {
    v16bf  v;
    __bf16 h[16];
    uint4  q[2];
};

// ---- async global->LDS 16B copy (per lane), fallback = regular copy -------
__device__ __forceinline__ void cp_async_b128(const __bf16* gsrc, __bf16* ldst) {
#if ASYNC_LDS
    __builtin_amdgcn_global_load_async_to_lds_b128((v4i*)gsrc, (v4i*)ldst, 0, 0);
#else
    *(uint4*)ldst = *(const uint4*)gsrc;
#endif
}

__device__ __forceinline__ void wait_async() {
#if ASYNC_LDS
#if __has_builtin(__builtin_amdgcn_s_wait_asynccnt)
    __builtin_amdgcn_s_wait_asynccnt(0);
#else
    asm volatile("s_wait_asynccnt 0" ::: "memory");
#endif
#endif
}

// A-fragment (16x32 bf16, M x K): lane = row (l&15); lanes 0-15 hold K {0-7,16-23},
// lanes 16-31 hold K {8-15,24-31}. Two contiguous 16B loads per lane.
__device__ __forceinline__ v16bf load_a_frag(const __bf16* base, int ld) {
    const int l  = threadIdx.x & 31;
    const int hi = l >> 4;
    const __bf16* p = base + (l & 15) * ld + hi * 8;
    BF16Frag f;
    f.q[0] = *(const uint4*)(p);
    f.q[1] = *(const uint4*)(p + 16);
    return f.v;
}

// B-fragment (32x16 bf16, K x N) from a source stored as [n][k] (column of B
// contiguous along K): lane = column (l&15); lanes 0-15 hold K {0-7,16-23},
// lanes 16-31 hold K {8-15,24-31}.
__device__ __forceinline__ v16bf load_b_frag(const __bf16* base, int ld) {
    const int l  = threadIdx.x & 31;
    const int hi = l >> 4;
    const __bf16* p = base + (l & 15) * ld + hi * 8;
    BF16Frag f;
    f.q[0] = *(const uint4*)(p);
    f.q[1] = *(const uint4*)(p + 16);
    return f.v;
}

__device__ __forceinline__ v8f wmma_bf16(v16bf a, v16bf b, v8f c) {
    return __builtin_amdgcn_wmma_f32_16x16x32_bf16(false, a, false, b,
                                                   (short)0, c, false, false);
}

// ---------------------------------------------------------------- convert
__global__ void f32_to_bf16_kernel(const float* __restrict__ in,
                                   __bf16* __restrict__ out, int n) {
    int i = blockIdx.x * blockDim.x + threadIdx.x;
    if (i < n) out[i] = (__bf16)in[i];
}

// ---------------------------------------------------------------- QKV GEMM
// C[8192,2304] = X[8192,768] * W[2304,768]^T   (bf16 in, bf16 out, f32 acc)
// Block tile 128x128, 8 waves as 2(m) x 4(n), wave tile 64x32 (4x2 WMMA,
// 64 accumulator VGPRs -> no spills).
__global__ void qkv_gemm_kernel(const __bf16* __restrict__ X,
                                const __bf16* __restrict__ W,
                                __bf16* __restrict__ QKV) {
    __shared__ __bf16 As[128 * 32];   // 8 KB
    __shared__ __bf16 Bs[128 * 32];   // 8 KB
    const int o0 = blockIdx.x * 128;
    const int m0 = blockIdx.y * 128;
    const int t  = threadIdx.x;
    const int wave = t >> 5, l = t & 31, lo = l & 15, hi = l >> 4;
    const int waveM = (wave >> 2) * 64, waveN = (wave & 3) * 32;
    v8f acc[4][2] = {};
    for (int kk = 0; kk < 768; kk += 32) {
        {   // A tile 128x32: thread t loads half a row (2 x 16B)
            int row = t >> 1, cb = (t & 1) * 16;
            const __bf16* src = X + (m0 + row) * 768 + kk + cb;
            __bf16* dst = As + row * 32 + cb;
            cp_async_b128(src,     dst);
            cp_async_b128(src + 8, dst + 8);
        }
        {   // B tile 128x32 (rows = output cols of W)
            int row = t >> 1, cb = (t & 1) * 16;
            const __bf16* src = W + (o0 + row) * 768 + kk + cb;
            __bf16* dst = Bs + row * 32 + cb;
            cp_async_b128(src,     dst);
            cp_async_b128(src + 8, dst + 8);
        }
        wait_async();
        __syncthreads();
        v16bf a[4], b;
        for (int i = 0; i < 4; ++i)
            a[i] = load_a_frag(As + (waveM + i * 16) * 32, 32);
        for (int j = 0; j < 2; ++j) {
            b = load_b_frag(Bs + (waveN + j * 16) * 32, 32);
            for (int i = 0; i < 4; ++i)
                acc[i][j] = wmma_bf16(a[i], b, acc[i][j]);
        }
        __syncthreads();
    }
    for (int i = 0; i < 4; ++i)
        for (int j = 0; j < 2; ++j)
            for (int r = 0; r < 8; ++r) {
                int m = m0 + waveM + i * 16 + r + hi * 8;
                int o = o0 + waveN + j * 16 + lo;
                QKV[m * 2304 + o] = (__bf16)acc[i][j][r];
            }
}

// ---------------------------------------------------------------- Attention
// Flash-style: per (b,head) [blockIdx.y], 128 query rows per block, key tiles
// of 32.  QKV layout: row m = b*1024+n, cols [0,768)=Q [768,1536)=K [1536,2304)=V,
// per-head slice = head*64 .. +64.
__global__ void attention_kernel(const __bf16* __restrict__ QKV,
                                 __bf16* __restrict__ O) {
    __shared__ __bf16 Ks[32 * 64];        // K tile row-major [key][d]
    __shared__ __bf16 Vt[64 * 32];        // V tile transposed [d][key]
    __shared__ __bf16 Ps[8 * 16 * 32];    // per-wave P tiles
    const int bh = blockIdx.y;
    const int b = bh / 12, head = bh % 12;
    const int t = threadIdx.x;
    const int wave = t >> 5, l = t & 31, lo = l & 15, hi = l >> 4;
    const int q0 = blockIdx.x * 128 + wave * 16;
    const int rowBase = b * 1024;
    const int colH = head * 64;

    v16bf qa0 = load_a_frag(QKV + (rowBase + q0) * 2304 + colH, 2304);       // d 0-31
    v16bf qa1 = load_a_frag(QKV + (rowBase + q0) * 2304 + colH + 32, 2304);  // d 32-63

    v8f o_acc[4] = {};
    float m_run[8], l_run[8];
    for (int r = 0; r < 8; ++r) { m_run[r] = -1e30f; l_run[r] = 0.f; }
    __bf16* Pw = Ps + wave * (16 * 32);

    for (int kt = 0; kt < 1024; kt += 32) {
        __syncthreads();
        {   // cooperative load: K via async-to-LDS, V transposed via VALU path
            int row = t >> 3, cb = (t & 7) * 8;
            const __bf16* ksrc = QKV + (rowBase + kt + row) * 2304 + 768 + colH + cb;
            cp_async_b128(ksrc, Ks + row * 64 + cb);
            const __bf16* vsrc = QKV + (rowBase + kt + row) * 2304 + 1536 + colH + cb;
            BF16Frag vf;
            vf.q[0] = *(const uint4*)vsrc;
            for (int e = 0; e < 8; ++e) Vt[(cb + e) * 32 + row] = vf.h[e];
        }
        wait_async();
        __syncthreads();

        // S = Q * K^T for 32 keys (two 16-key column groups, K-dim d=64)
        v8f s0 = {}, s1 = {};
        s0 = wmma_bf16(qa0, load_b_frag(Ks +  0 * 64 +  0, 64), s0);
        s0 = wmma_bf16(qa1, load_b_frag(Ks +  0 * 64 + 32, 64), s0);
        s1 = wmma_bf16(qa0, load_b_frag(Ks + 16 * 64 +  0, 64), s1);
        s1 = wmma_bf16(qa1, load_b_frag(Ks + 16 * 64 + 32, 64), s1);

        // Online softmax.  D-layout: lane holds col (l&15), rows r+8*hi.
        float fac[8], p0[8], p1[8];
        for (int r = 0; r < 8; ++r) {
            float sa = s0[r] * 0.125f;
            float sb = s1[r] * 0.125f;
            float mx = fmaxf(sa, sb);
            for (int off = 1; off < 16; off <<= 1)
                mx = fmaxf(mx, __shfl_xor(mx, off, 32));
            float m_new = fmaxf(m_run[r], mx);
            fac[r] = __expf(m_run[r] - m_new);
            p0[r]  = __expf(sa - m_new);
            p1[r]  = __expf(sb - m_new);
            float sum = p0[r] + p1[r];
            for (int off = 1; off < 16; off <<= 1)
                sum += __shfl_xor(sum, off, 32);
            l_run[r] = l_run[r] * fac[r] + sum;
            m_run[r] = m_new;
        }
        for (int c = 0; c < 4; ++c)
            for (int r = 0; r < 8; ++r)
                o_acc[c][r] *= fac[r];

        // P (D-layout) -> per-wave LDS -> reload as A-fragment (16x32)
        for (int r = 0; r < 8; ++r) {
            int row = r + hi * 8;
            Pw[row * 32 + lo]      = (__bf16)p0[r];
            Pw[row * 32 + 16 + lo] = (__bf16)p1[r];
        }
        asm volatile("s_wait_dscnt 0" ::: "memory");
        v16bf pa = load_a_frag(Pw, 32);

        // O += P * V  (K=32 keys, 4 column groups over d=64)
        for (int c = 0; c < 4; ++c)
            o_acc[c] = wmma_bf16(pa, load_b_frag(Vt + (c * 16) * 32, 32), o_acc[c]);
    }

    for (int r = 0; r < 8; ++r) l_run[r] = 1.f / l_run[r];
    for (int c = 0; c < 4; ++c)
        for (int r = 0; r < 8; ++r) {
            int m = rowBase + q0 + r + hi * 8;
            O[m * 768 + colH + c * 16 + lo] = (__bf16)(o_acc[c][r] * l_run[r]);
        }
}

// ---------------------------------------------------------------- Proj GEMM
// out[8192,768] = A[8192,768] * W[768,768]^T + bias   (f32 out)
// Same 128x128 / 64x32 tiling as the QKV GEMM.
__global__ void proj_gemm_kernel(const __bf16* __restrict__ A,
                                 const __bf16* __restrict__ W,
                                 const float* __restrict__ bias,
                                 float* __restrict__ out) {
    __shared__ __bf16 As[128 * 32];
    __shared__ __bf16 Bs[128 * 32];
    const int o0 = blockIdx.x * 128;
    const int m0 = blockIdx.y * 128;
    const int t  = threadIdx.x;
    const int wave = t >> 5, l = t & 31, lo = l & 15, hi = l >> 4;
    const int waveM = (wave >> 2) * 64, waveN = (wave & 3) * 32;
    v8f acc[4][2] = {};
    for (int kk = 0; kk < 768; kk += 32) {
        {
            int row = t >> 1, cb = (t & 1) * 16;
            const __bf16* src = A + (m0 + row) * 768 + kk + cb;
            __bf16* dst = As + row * 32 + cb;
            cp_async_b128(src,     dst);
            cp_async_b128(src + 8, dst + 8);
        }
        {
            int row = t >> 1, cb = (t & 1) * 16;
            const __bf16* src = W + (o0 + row) * 768 + kk + cb;
            __bf16* dst = Bs + row * 32 + cb;
            cp_async_b128(src,     dst);
            cp_async_b128(src + 8, dst + 8);
        }
        wait_async();
        __syncthreads();
        v16bf a[4], b;
        for (int i = 0; i < 4; ++i)
            a[i] = load_a_frag(As + (waveM + i * 16) * 32, 32);
        for (int j = 0; j < 2; ++j) {
            b = load_b_frag(Bs + (waveN + j * 16) * 32, 32);
            for (int i = 0; i < 4; ++i)
                acc[i][j] = wmma_bf16(a[i], b, acc[i][j]);
        }
        __syncthreads();
    }
    for (int i = 0; i < 4; ++i)
        for (int j = 0; j < 2; ++j)
            for (int r = 0; r < 8; ++r) {
                int m = m0 + waveM + i * 16 + r + hi * 8;
                int o = o0 + waveN + j * 16 + lo;
                out[m * 768 + o] = acc[i][j][r] + bias[o];
            }
}

// ---------------------------------------------------------------- launcher
extern "C" void kernel_launch(void* const* d_in, const int* in_sizes, int n_in,
                              void* d_out, int out_size, void* d_ws, size_t ws_size,
                              hipStream_t stream) {
    (void)in_sizes; (void)n_in; (void)out_size; (void)ws_size;
    const float* x      = (const float*)d_in[0];   // 8*32*32*768
    const float* qkv_w  = (const float*)d_in[1];   // 2304*768
    const float* proj_w = (const float*)d_in[2];   // 768*768
    const float* proj_b = (const float*)d_in[3];   // 768
    float* out = (float*)d_out;

    const int NX = 8 * 32 * 32 * 768;   // 6291456
    const int NW = 2304 * 768;          // 1769472
    const int NP = 768 * 768;           // 589824

    __bf16* Xb    = (__bf16*)d_ws;
    __bf16* Wqkv  = Xb + NX;
    __bf16* Wproj = Wqkv + NW;
    __bf16* QKV   = Wproj + NP;              // 8192*2304 bf16
    __bf16* AttnO = QKV + 8192 * 2304;       // 8192*768 bf16

    f32_to_bf16_kernel<<<(NX + 255) / 256, 256, 0, stream>>>(x, Xb, NX);
    f32_to_bf16_kernel<<<(NW + 255) / 256, 256, 0, stream>>>(qkv_w, Wqkv, NW);
    f32_to_bf16_kernel<<<(NP + 255) / 256, 256, 0, stream>>>(proj_w, Wproj, NP);

    qkv_gemm_kernel<<<dim3(18, 64), 256, 0, stream>>>(Xb, Wqkv, QKV);
    attention_kernel<<<dim3(8, 96), 256, 0, stream>>>(QKV, AttnO);
    proj_gemm_kernel<<<dim3(6, 64), 256, 0, stream>>>(AttnO, Wproj, proj_b, out);
}
